// SlowButCorrectMultiheadAttention_89515708383409
// MI455X (gfx1250) — matrix operand order repaired and checked by reading, not compiled, not run on hardware
//
#include <hip/hip_runtime.h>
#include <hip/hip_bf16.h>

// ---------- types ----------
typedef __attribute__((ext_vector_type(16))) __bf16 v16bf;
typedef __attribute__((ext_vector_type(8)))  __bf16 v8bf;
typedef __attribute__((ext_vector_type(8)))  float  v8f;
typedef unsigned short u16;
typedef unsigned int   u32;

// ---------- problem constants ----------
constexpr int S_ = 2048;
constexpr int B_ = 4;
constexpr int E_ = 1024;
constexpr int H_ = 16;
constexpr int D_ = 64;
constexpr int NT = S_ * B_;   // 8192 tokens

// ---------- helpers ----------
__device__ __forceinline__ u16 f2bf(float f) {
    u32 u = __builtin_bit_cast(u32, f);
    u32 r = u + 0x7FFFu + ((u >> 16) & 1u);   // round-to-nearest-even
    return (u16)(r >> 16);
}

__device__ __forceinline__ v8bf ld8(const u16* p) {
    return *(const v8bf*)p;
}

__device__ __forceinline__ v16bf cat8(v8bf lo, v8bf hi) {
    return __builtin_shufflevector(lo, hi, 0,1,2,3,4,5,6,7,8,9,10,11,12,13,14,15);
}

// A-fragment (16x32 bf16, M x K): lane row = lane%16.
// lanes 0-15 hold K={0..7,16..23}; lanes 16-31 hold K={8..15,24..31}.
__device__ __forceinline__ v16bf loadA(const u16* row, int hi) {
    const int o = hi ? 8 : 0;
    return cat8(ld8(row + o), ld8(row + 16 + o));
}

// B-fragment (32x16 bf16, K x N): lane column = lane%16.
// lanes 0-15 hold K={0..7,16..23}; lanes 16-31 hold K={8..15,24..31}.
__device__ __forceinline__ v16bf loadB(const u16* col, int hi) {
    const u16* p = col + (hi ? 8 : 0);
    return cat8(ld8(p), ld8(p + 16));
}

__device__ __forceinline__ v8f wmma_bf16(v16bf a, v16bf b, v8f c) {
    return __builtin_amdgcn_wmma_f32_16x16x32_bf16(
        /*neg_a=*/false, a, /*neg_b=*/false, b,
        /*c_mod=*/(short)0, c, /*reuse_a=*/false, /*reuse_b=*/false);
}

// ---------- fp32 -> bf16 conversion ----------
__global__ __launch_bounds__(256) void cvt_bf16_kernel(const float* __restrict__ in,
                                                       u16* __restrict__ out, int n) {
    int i = blockIdx.x * 256 + threadIdx.x;
    if (i < n) out[i] = f2bf(in[i]);
}

// ---------- GEMM: Y[m,n] = sum_k A[m,k] * W[n,k] + bias[n] ----------
// A: [NT, E] bf16 row-major.  W: [E, E] bf16 row-major (rows = output features).
// Each wave computes a 32x64 output block (2 m-tiles x 4 n-tiles): every weight
// B-fragment is reused by two WMMAs.
// OUT_MODE: 0 = bf16 token-major, 1 = bf16 transposed into Vt[b,h,d,s],
//           2 = fp32 token-major, 3 = bf16 token-major pre-scaled by 1/sqrt(D)
template <int OUT_MODE>
__global__ __launch_bounds__(128) void gemm_kernel(const u16* __restrict__ A,
                                                   const u16* __restrict__ W,
                                                   const float* __restrict__ bias,
                                                   u16* __restrict__ Yb,
                                                   float* __restrict__ Yf) {
    const int wid  = (blockIdx.x * 128 + threadIdx.x) >> 5;   // global wave id
    const int lane = threadIdx.x & 31;
    const int hi   = lane >> 4;
    const int l16  = lane & 15;

    const int mt  = wid >> 4;   // 256 row blocks of 32
    const int n64 = wid & 15;   // 16 column strips of 64

    const u16* arow0 = A + (size_t)(mt * 32 + l16) * E_;
    const u16* arow1 = arow0 + (size_t)16 * E_;
    const u16* wrow0 = W + (size_t)(n64 * 64 + l16) * E_;

    v8f acc[2][4] = {};
    for (int k = 0; k < E_; k += 32) {
        v16bf af0 = loadA(arow0 + k, hi);
        v16bf af1 = loadA(arow1 + k, hi);
#pragma unroll
        for (int nt = 0; nt < 4; ++nt) {
            v16bf bf = loadB(wrow0 + (size_t)(nt * 16) * E_ + k, hi);
            acc[0][nt] = wmma_bf16(af0, bf, acc[0][nt]);
            acc[1][nt] = wmma_bf16(af1, bf, acc[1][nt]);
        }
    }

#pragma unroll
    for (int mi = 0; mi < 2; ++mi) {
#pragma unroll
        for (int nt = 0; nt < 4; ++nt) {
            const int ncol = n64 * 64 + nt * 16 + l16;
            const float bv = bias[ncol];
#pragma unroll
            for (int j = 0; j < 8; ++j) {
                const int m = mt * 32 + mi * 16 + j + 8 * hi;   // token index
                const float val = acc[mi][nt][j] + bv;
                if (OUT_MODE == 2) {
                    Yf[(size_t)m * E_ + ncol] = val;
                } else if (OUT_MODE == 0) {
                    Yb[(size_t)m * E_ + ncol] = f2bf(val);
                } else if (OUT_MODE == 3) {
                    Yb[(size_t)m * E_ + ncol] = f2bf(val * 0.125f);  // 1/sqrt(64)
                } else {
                    // Vt[(b*H+h)*D + d][s];  ncol = h*64+d, token m = s*B + b
                    const int h = ncol >> 6, d = ncol & 63;
                    const int s = m >> 2,  b = m & 3;
                    Yb[(size_t)((b * H_ + h) * D_ + d) * S_ + s] = f2bf(val);
                }
            }
        }
    }
}

// ---------- Flash attention: one wave per (b, h, 32-query block) ----------
// Transposed score tiles S^T = K @ Q^T: exp(S^T)'s C-layout (lane = query,
// VGPR = key) is directly the A-operand layout for P@V — no cross-lane moves.
// Each K A-fragment and V B-fragment feeds two WMMAs (two query tiles).
__global__ __launch_bounds__(32) void attn_kernel(const u16* __restrict__ Q,   // [NT,E], pre-scaled
                                                  const u16* __restrict__ K,   // [NT,E]
                                                  const u16* __restrict__ Vt,  // [B*H*D, S]
                                                  u16* __restrict__ O) {       // [NT,E]
    const int lane = threadIdx.x & 31;
    const int hi   = lane >> 4;
    const int l16  = lane & 15;

    const int id = blockIdx.x;         // B*H*(S/32) = 4096
    const int qt = id & 63;            // 32-query block
    const int bh = id >> 6;            // b*H + h
    const int b  = bh >> 4;
    const int h  = bh & 15;

    // Q as B-operand (column q = Q row, contiguous over d); constant over key loop
    v16bf bq[2][2];
#pragma unroll
    for (int qi = 0; qi < 2; ++qi) {
        const int qtok = (qt * 32 + qi * 16 + l16) * B_ + b;
        const u16* qrow = Q + (size_t)qtok * E_ + h * 64;
        bq[qi][0] = loadB(qrow, hi);        // d = 0..31
        bq[qi][1] = loadB(qrow + 32, hi);   // d = 32..63
    }

    float mrun[2] = { -1e30f, -1e30f };
    float lrun[2] = { 0.0f, 0.0f };
    v8f acc[2][4] = {};                // per query tile: O tile 16x64

    for (int kb = 0; kb < S_; kb += 32) {
        // --- K A-fragments (shared by both query tiles) ---
        const u16* krow = K + (size_t)((kb + l16) * B_ + b) * E_ + h * 64;
        const u16* krow1 = krow + (size_t)16 * B_ * E_;   // keys +16
        const v16bf ka0a = loadA(krow, hi);
        const v16bf ka0b = loadA(krow + 32, hi);
        const v16bf ka1a = loadA(krow1, hi);
        const v16bf ka1b = loadA(krow1 + 32, hi);

        v16bf af[2];
#pragma unroll
        for (int qi = 0; qi < 2; ++qi) {
            // transposed score tiles: M = key, N = query, contraction over d
            v8f s0 = {}, s1 = {};
            s0 = wmma_bf16(ka0a, bq[qi][0], s0);
            s0 = wmma_bf16(ka0b, bq[qi][1], s0);
            s1 = wmma_bf16(ka1a, bq[qi][0], s1);
            s1 = wmma_bf16(ka1b, bq[qi][1], s1);

            // online softmax; each lane owns query l16 (keys j+8*hi per tile)
            float tmax = -1e30f;
#pragma unroll
            for (int j = 0; j < 8; ++j)
                tmax = fmaxf(tmax, fmaxf(s0[j], s1[j]));
            tmax = fmaxf(tmax, __shfl_xor(tmax, 16, 32));
            const float mnew  = fmaxf(mrun[qi], tmax);
            const float scale = __expf(mrun[qi] - mnew);

            float psum = 0.0f;
#pragma unroll
            for (int j = 0; j < 8; ++j) {
                const float p0 = __expf(s0[j] - mnew);
                const float p1 = __expf(s1[j] - mnew);
                psum += p0 + p1;
                af[qi][j]     = __builtin_bit_cast(__bf16, f2bf(p0));  // K=(hi?8:0)+j
                af[qi][8 + j] = __builtin_bit_cast(__bf16, f2bf(p1));  // K=16+(hi?8:0)+j
            }
            psum += __shfl_xor(psum, 16, 32);
            lrun[qi] = lrun[qi] * scale + psum;
            mrun[qi] = mnew;

            // rescale accumulators: factor indexed per C-row (query j+8*hi)
#pragma unroll
            for (int j = 0; j < 8; ++j) {
                const float f = __shfl(scale, j + 8 * hi, 32);
#pragma unroll
                for (int n = 0; n < 4; ++n) acc[qi][n][j] *= f;
            }
        }

        // --- P @ V : each V B-fragment feeds both query tiles ---
#pragma unroll
        for (int n = 0; n < 4; ++n) {
            const int d = n * 16 + l16;
            const u16* vrow = Vt + (size_t)(bh * D_ + d) * S_ + kb + (hi ? 8 : 0);
            const v16bf bvf = cat8(ld8(vrow), ld8(vrow + 16));
            acc[0][n] = wmma_bf16(af[0], bvf, acc[0][n]);
            acc[1][n] = wmma_bf16(af[1], bvf, acc[1][n]);
        }
    }

    // --- finalize: divide by softmax denominator and store ---
#pragma unroll
    for (int qi = 0; qi < 2; ++qi) {
#pragma unroll
        for (int j = 0; j < 8; ++j) {
            const float inv = 1.0f / __shfl(lrun[qi], j + 8 * hi, 32);
            const int tok = (qt * 32 + qi * 16 + j + 8 * hi) * B_ + b;
#pragma unroll
            for (int n = 0; n < 4; ++n) {
                O[(size_t)tok * E_ + h * 64 + n * 16 + l16] = f2bf(acc[qi][n][j] * inv);
            }
        }
    }
}

// ---------- launch ----------
extern "C" void kernel_launch(void* const* d_in, const int* in_sizes, int n_in,
                              void* d_out, int out_size, void* d_ws, size_t ws_size,
                              hipStream_t stream) {
    const float* x  = (const float*)d_in[0];
    const float* Wq = (const float*)d_in[1];
    const float* bq = (const float*)d_in[2];
    const float* Wk = (const float*)d_in[3];
    const float* bk = (const float*)d_in[4];
    const float* Wv = (const float*)d_in[5];
    const float* bv = (const float*)d_in[6];
    const float* Wo = (const float*)d_in[7];
    const float* bo = (const float*)d_in[8];
    float* out = (float*)d_out;

    // workspace carve (bf16 elements)
    u16* xb  = (u16*)d_ws;               // NT*E
    u16* wqb = xb  + (size_t)NT * E_;    // E*E
    u16* wkb = wqb + (size_t)E_ * E_;
    u16* wvb = wkb + (size_t)E_ * E_;
    u16* wob = wvb + (size_t)E_ * E_;
    u16* Qb  = wob + (size_t)E_ * E_;    // NT*E
    u16* Kb  = Qb  + (size_t)NT * E_;
    u16* Vtb = Kb  + (size_t)NT * E_;    // [B*H*D, S]
    u16* Ob  = Vtb + (size_t)NT * E_;

    // fp32 -> bf16 conversions
    {
        const int n = NT * E_;
        cvt_bf16_kernel<<<(n + 255) / 256, 256, 0, stream>>>(x, xb, n);
        const int nw = E_ * E_;
        cvt_bf16_kernel<<<(nw + 255) / 256, 256, 0, stream>>>(Wq, wqb, nw);
        cvt_bf16_kernel<<<(nw + 255) / 256, 256, 0, stream>>>(Wk, wkb, nw);
        cvt_bf16_kernel<<<(nw + 255) / 256, 256, 0, stream>>>(Wv, wvb, nw);
        cvt_bf16_kernel<<<(nw + 255) / 256, 256, 0, stream>>>(Wo, wob, nw);
    }

    // Q, K, V projections (256 m-blocks * 16 n-strips = 4096 waves, 4 waves/block)
    const dim3 ggrid(1024), gblk(128);
    gemm_kernel<3><<<ggrid, gblk, 0, stream>>>(xb, wqb, bq, Qb, nullptr);   // Q pre-scaled
    gemm_kernel<0><<<ggrid, gblk, 0, stream>>>(xb, wkb, bk, Kb, nullptr);
    gemm_kernel<1><<<ggrid, gblk, 0, stream>>>(xb, wvb, bv, Vtb, nullptr);  // V transposed

    // attention: one wave per (b, h, 32-query block)
    attn_kernel<<<B_ * H_ * (S_ / 32), 32, 0, stream>>>(Qb, Kb, Vtb, Ob);

    // output projection, fp32 result
    gemm_kernel<2><<<ggrid, gblk, 0, stream>>>(Ob, wob, bo, nullptr, out);
}
